// GptOssExperts_63307817943053
// MI455X (gfx1250) — matrix operand order repaired and checked by reading, not compiled, not run on hardware
//
#include <hip/hip_runtime.h>

// ---------------------------------------------------------------------------
// GPT-OSS MoE experts forward for MI455X (gfx1250), wave32 + WMMA bf16.
//   E=8, H=2048, I=2048, T=2048
// Pass 1: inter[e,t,i] = act( X @ Wgu[e] + bgu[e] )   (bf16 intermediate in ws)
// Pass 2: out[t,h]     = sum_e r[t,e] * ( inter[e] @ Wd[e] + bd[e] )
// Memory-bound (~400MB f32 weights, ~19us @ 23.3TB/s); math done in bf16 WMMA
// with in-register f32->bf16 conversion of the weight stream. Kernel 2 stages
// its (already-bf16) A tile with GLOBAL_LOAD_ASYNC_TO_LDS_B128 (ASYNCcnt).
// ---------------------------------------------------------------------------

typedef __attribute__((ext_vector_type(16))) __bf16 v16bf;
typedef __attribute__((ext_vector_type(2)))  __bf16 v2bf;
typedef __attribute__((ext_vector_type(8)))  float  v8f;

#define E_    8
#define H_    2048
#define I_    2048
#define T_    2048
#define TWOI_ 4096

#define BM  128   // token rows per block
#define BC  64    // output channels (or h cols) per block
#define BK  32    // k-step (bf16 WMMA K)
#define PAD 40    // LDS row stride in halves (32 data + 8 pad -> 80B rows)

union FragBF { v16bf v; uint4 q[2]; };
union Pack8  { __bf16 h[8]; uint4 q; };

__device__ __forceinline__ v8f vzero8() {
  v8f z = {0.f, 0.f, 0.f, 0.f, 0.f, 0.f, 0.f, 0.f};
  return z;
}

// ---------------------------------------------------------------------------
// Kernel 1: gate_up GEMM + bias + clamp + glu  ->  inter (bf16)
// Block tile: 128 tokens x 64 channels (=128 interleaved gu columns).
// B staged deinterleaved+transposed in LDS: Bg/Bu[c][k], so the gate/up pair
// for a channel lands in the SAME lane of matching accumulators (no shuffles).
// ---------------------------------------------------------------------------
__global__ __launch_bounds__(256) void gateup_act_kernel(
    const float* __restrict__ X,     // (T,H)
    const float* __restrict__ W,     // (E,H,2I)
    const float* __restrict__ Bias,  // (E,2I)
    __bf16* __restrict__ inter)      // (E,T,I) bf16
{
  __shared__ __attribute__((aligned(16))) __bf16 As[BM * PAD];
  __shared__ __attribute__((aligned(16))) __bf16 Bg[BC * PAD];
  __shared__ __attribute__((aligned(16))) __bf16 Bu[BC * PAD];

  const int c0 = blockIdx.x * BC;   // channel base (inter channel index)
  const int t0 = blockIdx.y * BM;
  const int e  = blockIdx.z;
  const float* We = W + (size_t)e * H_ * TWOI_;

  const int tid  = threadIdx.x;
  const int lane = tid & 31;
  const int wid  = tid >> 5;   // 8 waves
  const int ct   = wid & 3;    // channel tile (16 channels)
  const int mh   = wid >> 2;   // token half (64 rows)

  v8f accg[4], accu[4];
#pragma unroll
  for (int i = 0; i < 4; ++i) { accg[i] = vzero8(); accu[i] = vzero8(); }

  for (int k0 = 0; k0 < H_; k0 += BK) {
    __syncthreads();
    // ---- stage A: hidden tile (128 x 32) f32 -> bf16, row-major [m][k]
#pragma unroll
    for (int it = 0; it < 2; ++it) {
      const int m  = (tid >> 2) + it * 64;
      const int ks = (tid & 3) * 8;
      const float* src = X + (size_t)(t0 + m) * H_ + k0 + ks;
      float4 f0 = *(const float4*)src;
      float4 f1 = *(const float4*)(src + 4);
      Pack8 p;
      p.h[0] = (__bf16)f0.x; p.h[1] = (__bf16)f0.y;
      p.h[2] = (__bf16)f0.z; p.h[3] = (__bf16)f0.w;
      p.h[4] = (__bf16)f1.x; p.h[5] = (__bf16)f1.y;
      p.h[6] = (__bf16)f1.z; p.h[7] = (__bf16)f1.w;
      *(uint4*)&As[m * PAD + ks] = p.q;
    }
    // ---- stage B: weight tile (32 x 128 gu-cols) f32 -> bf16,
    //      transposed + deinterleaved into Bg/Bu [c][k] (k pair-packed b32)
#pragma unroll
    for (int it = 0; it < 2; ++it) {
      const int k  = (((tid >> 5) + it * 8) << 1);  // even k row
      const int j4 = (tid & 31) * 4;                // gu-col offset in tile
      const float* src = We + (size_t)(k0 + k) * TWOI_ + 2 * c0 + j4;
      __builtin_prefetch(src + (size_t)BK * TWOI_, 0, 0);  // next k-tile
      float4 r0 = *(const float4*)src;           // gate(c),up(c),gate(c+1),up(c+1)
      float4 r1 = *(const float4*)(src + TWOI_); // same cols, row k+1
      const int c = j4 >> 1;
      v2bf g0 = {(__bf16)r0.x, (__bf16)r1.x};
      v2bf u0 = {(__bf16)r0.y, (__bf16)r1.y};
      v2bf g1 = {(__bf16)r0.z, (__bf16)r1.z};
      v2bf u1 = {(__bf16)r0.w, (__bf16)r1.w};
      *(unsigned*)&Bg[(c + 0) * PAD + k] = __builtin_bit_cast(unsigned, g0);
      *(unsigned*)&Bu[(c + 0) * PAD + k] = __builtin_bit_cast(unsigned, u0);
      *(unsigned*)&Bg[(c + 1) * PAD + k] = __builtin_bit_cast(unsigned, g1);
      *(unsigned*)&Bu[(c + 1) * PAD + k] = __builtin_bit_cast(unsigned, u1);
    }
    __syncthreads();

    // ---- B fragments (32x16): lanes 0-15 K=0..15, lanes 16-31 K=16..31
    FragBF fbg, fbu;
    {
      const int n  = ct * 16 + (lane & 15);
      const int kh = (lane >> 4) * 16;
      fbg.q[0] = *(const uint4*)&Bg[n * PAD + kh];
      fbg.q[1] = *(const uint4*)&Bg[n * PAD + kh + 8];
      fbu.q[0] = *(const uint4*)&Bu[n * PAD + kh];
      fbu.q[1] = *(const uint4*)&Bu[n * PAD + kh + 8];
    }
    // ---- A fragments (16x32): lane<16 K=0..15, lane>=16 K=16..31
#pragma unroll
    for (int i = 0; i < 4; ++i) {
      FragBF fa;
      const int m  = mh * 64 + i * 16 + (lane & 15);
      const int kh = (lane >> 4) * 16;
      fa.q[0] = *(const uint4*)&As[m * PAD + kh];
      fa.q[1] = *(const uint4*)&As[m * PAD + kh + 8];
      accg[i] = __builtin_amdgcn_wmma_f32_16x16x32_bf16(
          false, fa.v, false, fbg.v, (short)0, accg[i], false, false);
      accu[i] = __builtin_amdgcn_wmma_f32_16x16x32_bf16(
          false, fa.v, false, fbu.v, (short)0, accu[i], false, false);
    }
  }

  // ---- epilogue: bias, clamp, glu = g*sigmoid(1.702g), inter = (u+1)*glu
  const int c = c0 + ct * 16 + (lane & 15);
  const float2 bia = *(const float2*)(Bias + (size_t)e * TWOI_ + 2 * c);
#pragma unroll
  for (int i = 0; i < 4; ++i) {
#pragma unroll
    for (int r = 0; r < 8; ++r) {
      const int t = t0 + mh * 64 + i * 16 + r + ((lane >> 4) << 3);
      float g = accg[i][r] + bia.x;
      float u = accu[i][r] + bia.y;
      g = fminf(g, 7.0f);
      u = fminf(fmaxf(u, -7.0f), 7.0f);
      const float glu = g / (1.0f + __expf(-1.702f * g));
      inter[((size_t)e * T_ + t) * I_ + c] = (__bf16)((u + 1.0f) * glu);
    }
  }
}

// ---------------------------------------------------------------------------
// Kernel 2: per-expert down GEMM, fold routing-weighted sum across experts in
// registers, add routed bias, write each out element exactly once (no atomics).
// Block tile: 128 tokens x 64 h-cols; loops over all 8 experts.
// A tile staged via async global->LDS DMA (GLOBAL_LOAD_ASYNC_TO_LDS_B128).
// ---------------------------------------------------------------------------
__global__ __launch_bounds__(256) void down_route_kernel(
    const __bf16* __restrict__ inter, // (E,T,I) bf16
    const float*  __restrict__ Wd,    // (E,I,H)
    const float*  __restrict__ DBias, // (E,H)
    const float*  __restrict__ Rout,  // (T,E)
    float* __restrict__ out)          // (T,H)
{
  __shared__ __attribute__((aligned(16))) __bf16 As[BM * PAD];
  __shared__ __attribute__((aligned(16))) __bf16 Bs[BC * PAD];

  const int h0 = blockIdx.x * BC;
  const int t0 = blockIdx.y * BM;

  const int tid  = threadIdx.x;
  const int lane = tid & 31;
  const int wid  = tid >> 5;
  const int nt   = wid & 3;    // h tile (16 cols)
  const int mh   = wid >> 2;   // token half (64 rows)

  v8f fin[4];
#pragma unroll
  for (int i = 0; i < 4; ++i) fin[i] = vzero8();

  for (int e = 0; e < E_; ++e) {
    const __bf16* Ae = inter + (size_t)e * T_ * I_;
    const float*  We = Wd    + (size_t)e * I_ * H_;

    v8f ce[4];
#pragma unroll
    for (int i = 0; i < 4; ++i) ce[i] = vzero8();

    for (int k0 = 0; k0 < I_; k0 += BK) {
      __syncthreads();
      // ---- stage A: inter tile (128 x 32) already bf16 — async DMA to LDS.
      // INST_OFFSET is added to both the LDS and global addresses (ISA 08 §4.4),
      // so the second 16B chunk is expressed purely with offset:16.
      {
        const int m  = tid >> 1;
        const int ks = (tid & 1) * 16;
        const __bf16* src = Ae + (size_t)(t0 + m) * I_ + k0 + ks;
        const unsigned lds0 =
            (unsigned)(unsigned long long)(&As[m * PAD + ks]);  // flat->LDS off
        asm volatile("global_load_async_to_lds_b128 %0, %1, off"
                     :: "v"(lds0), "v"(src) : "memory");
        asm volatile("global_load_async_to_lds_b128 %0, %1, off offset:16"
                     :: "v"(lds0), "v"(src) : "memory");
      }
      // ---- stage B: down tile (32 x 64) f32 -> bf16 transposed [h][k]
#pragma unroll
      for (int it = 0; it < 2; ++it) {
        const int k  = (((tid >> 5) + it * 8) << 1);
        const int n2 = (tid & 31) * 2;
        const float* src = We + (size_t)(k0 + k) * H_ + h0 + n2;
        __builtin_prefetch(src + (size_t)BK * H_, 0, 0);  // next k-tile
        float2 r0 = *(const float2*)src;
        float2 r1 = *(const float2*)(src + H_);
        v2bf p0 = {(__bf16)r0.x, (__bf16)r1.x};
        v2bf p1 = {(__bf16)r0.y, (__bf16)r1.y};
        *(unsigned*)&Bs[(n2 + 0) * PAD + k] = __builtin_bit_cast(unsigned, p0);
        *(unsigned*)&Bs[(n2 + 1) * PAD + k] = __builtin_bit_cast(unsigned, p1);
      }
      // Drain this wave's async DMA before the workgroup barrier.
      asm volatile("s_wait_asynccnt 0" ::: "memory");
      __syncthreads();

      FragBF fb;
      {
        const int n  = nt * 16 + (lane & 15);
        const int kh = (lane >> 4) * 16;
        fb.q[0] = *(const uint4*)&Bs[n * PAD + kh];
        fb.q[1] = *(const uint4*)&Bs[n * PAD + kh + 8];
      }
#pragma unroll
      for (int i = 0; i < 4; ++i) {
        FragBF fa;
        const int m  = mh * 64 + i * 16 + (lane & 15);
        const int kh = (lane >> 4) * 16;
        fa.q[0] = *(const uint4*)&As[m * PAD + kh];
        fa.q[1] = *(const uint4*)&As[m * PAD + kh + 8];
        ce[i] = __builtin_amdgcn_wmma_f32_16x16x32_bf16(
            false, fa.v, false, fb.v, (short)0, ce[i], false, false);
      }
    }

    // ---- fold this expert: fin += r[t,e] * (ce + down_bias[e,h])
    const int hc = h0 + nt * 16 + (lane & 15);
    const float db = DBias[(size_t)e * H_ + hc];
#pragma unroll
    for (int i = 0; i < 4; ++i) {
#pragma unroll
      for (int r = 0; r < 8; ++r) {
        const int t = t0 + mh * 64 + i * 16 + r + ((lane >> 4) << 3);
        const float rw = Rout[(size_t)t * E_ + e];
        fin[i][r] = fmaf(rw, ce[i][r] + db, fin[i][r]);
      }
    }
  }

  const int hc = h0 + nt * 16 + (lane & 15);
#pragma unroll
  for (int i = 0; i < 4; ++i) {
#pragma unroll
    for (int r = 0; r < 8; ++r) {
      const int t = t0 + mh * 64 + i * 16 + r + ((lane >> 4) << 3);
      out[(size_t)t * H_ + hc] = fin[i][r];
    }
  }
}

// ---------------------------------------------------------------------------
extern "C" void kernel_launch(void* const* d_in, const int* in_sizes, int n_in,
                              void* d_out, int out_size, void* d_ws, size_t ws_size,
                              hipStream_t stream) {
  const float* X   = (const float*)d_in[0];  // hidden_states (T,H)
  const float* Rt  = (const float*)d_in[1];  // routing_weights (T,E)
  const float* Wgu = (const float*)d_in[2];  // gate_up_proj (E,H,2I)
  const float* Bgu = (const float*)d_in[3];  // gate_up_proj_bias (E,2I)
  const float* Wd  = (const float*)d_in[4];  // down_proj (E,I,H)
  const float* Bd  = (const float*)d_in[5];  // down_proj_bias (E,H)
  float* outp = (float*)d_out;               // (T,H) f32
  __bf16* inter = (__bf16*)d_ws;             // (E,T,I) bf16 = 64 MiB scratch

  dim3 g1(I_ / BC, T_ / BM, E_);   // (32, 16, 8)
  gateup_act_kernel<<<g1, 256, 0, stream>>>(X, Wgu, Bgu, inter);

  dim3 g2(H_ / BC, T_ / BM, 1);    // (32, 16)
  down_route_kernel<<<g2, 256, 0, stream>>>(inter, Wd, Bd, Rt, outp);

  (void)in_sizes; (void)n_in; (void)out_size; (void)ws_size;
}
